// SelfMultiheadAttention_1047972020976
// MI455X (gfx1250) — matrix operand order repaired and checked by reading, not compiled, not run on hardware
//
#include <hip/hip_runtime.h>

#define EMBED   512
#define HEADS   16
#define HD      32
#define NNODE   1024
#define NGRAPH  4
#define SCALING 0.17677669529663687f   // 32^-0.5

typedef __attribute__((ext_vector_type(16))) __bf16 v16bf;
typedef __attribute__((ext_vector_type(8)))  __bf16 v8bf;
typedef __attribute__((ext_vector_type(8)))  float  v8f;
typedef __attribute__((ext_vector_type(4)))  float  f32x4;

static __device__ __forceinline__ v8f wmma_bf16(v16bf a, v16bf b, v8f c) {
  // v_wmma_f32_16x16x32_bf16: (neg_a, A, neg_b, B, c_mod, C, reuse_a, reuse_b)
  return __builtin_amdgcn_wmma_f32_16x16x32_bf16(false, a, false, b, (short)0, c, false, false);
}

// A fragment (16x32 bf16) from an fp32 row: lane m = lane&15, half = lane>>4.
// elements [0..7]  = row[k0 + 16*half + 0..7]
// elements [8..15] = row[k0 + 16*half + 8 + 0..7]
static __device__ __forceinline__ v16bf load_a_f32(const float* row, int half) {
  const f32x4* p0 = (const f32x4*)(row + 16 * half);
  const f32x4* p1 = (const f32x4*)(row + 16 * half + 8);
  f32x4 c0 = p0[0], c1 = p0[1], c2 = p1[0], c3 = p1[1];
  v16bf r;
#pragma unroll
  for (int i = 0; i < 4; ++i) {
    r[i]      = (__bf16)c0[i];
    r[4 + i]  = (__bf16)c1[i];
    r[8 + i]  = (__bf16)c2[i];
    r[12 + i] = (__bf16)c3[i];
  }
  return r;
}

// B fragment half-row: 16 contiguous fp32 -> 16 bf16 (K contiguous per half)
static __device__ __forceinline__ v16bf load_b16_f32(const float* p) {
  const f32x4* q = (const f32x4*)p;
  f32x4 c0 = q[0], c1 = q[1], c2 = q[2], c3 = q[3];
  v16bf r;
#pragma unroll
  for (int i = 0; i < 4; ++i) {
    r[i]      = (__bf16)c0[i];
    r[4 + i]  = (__bf16)c1[i];
    r[8 + i]  = (__bf16)c2[i];
    r[12 + i] = (__bf16)c3[i];
  }
  return r;
}

// ---------------------------------------------------------------------------
// Kernel 1: qkv = query @ W_in^T + b_in, scattered to Q (scaled) / K / V^T bf16
// M=4096, N=1536, K=512. One wave = 16x64 output tile (4 WMMA accumulators).
// ---------------------------------------------------------------------------
__global__ __launch_bounds__(256) void qkv_kernel(
    const float* __restrict__ query, const float* __restrict__ W_in,
    const float* __restrict__ b_in, __bf16* __restrict__ Qb,
    __bf16* __restrict__ Kb, __bf16* __restrict__ Vt) {
  int wave = (blockIdx.x * blockDim.x + threadIdx.x) >> 5;
  int lane = threadIdx.x & 31;
  int half = lane >> 4, nl = lane & 15;
  int mt = wave / 24, nt = wave % 24;     // 256 x 24 wave tiles
  int m0 = mt * 16, o0 = nt * 64;

  v8f acc[4] = {};
  const float* arow = query + (size_t)(m0 + nl) * EMBED;
  for (int k0 = 0; k0 < EMBED; k0 += 32) {
    v16bf a = load_a_f32(arow + k0, half);
#pragma unroll
    for (int t = 0; t < 4; ++t) {
      int o = o0 + t * 16 + nl;
      v16bf b = load_b16_f32(W_in + (size_t)o * EMBED + k0 + 16 * half);
      acc[t] = wmma_bf16(a, b, acc[t]);
    }
  }
#pragma unroll
  for (int t = 0; t < 4; ++t) {
    int o = o0 + t * 16 + nl;
    float bias = b_in[o];
    int seg = o >> 9, within = o & 511;
    int h = within >> 5, d = within & 31;
#pragma unroll
    for (int v = 0; v < 8; ++v) {
      int r = m0 + v + 8 * half;          // token row = n*4 + g
      int nn = r >> 2, g = r & 3;
      int gh = g * 16 + h;
      float val = acc[t][v] + bias;
      if (seg == 0)
        Qb[((size_t)gh * NNODE + nn) * HD + d] = (__bf16)(val * SCALING);
      else if (seg == 1)
        Kb[((size_t)gh * NNODE + nn) * HD + d] = (__bf16)val;
      else
        Vt[((size_t)gh * HD + d) * NNODE + nn] = (__bf16)val;
    }
  }
}

// ---------------------------------------------------------------------------
// Kernel 2: fused flash attention per (g,h). One wave = 16 query rows,
// streams 32 key columns per step: 2 score WMMAs + bias + online softmax +
// P->LDS->A-frag + 2 P.V WMMAs. Bias is the 256MB one-shot stream -> loaded
// non-temporal (TH=NT) so the ~8MB K/V working set stays L2-resident.
// ---------------------------------------------------------------------------
__global__ __launch_bounds__(128) void attn_kernel(
    const float* __restrict__ bias, const __bf16* __restrict__ Qb,
    const __bf16* __restrict__ Kb, const __bf16* __restrict__ Vt,
    __bf16* __restrict__ attnbuf) {
  __shared__ __bf16 plds[4][16][32];      // 1KB per wave, wave-private
  int w = threadIdx.x >> 5;
  int lane = threadIdx.x & 31;
  int half = lane >> 4, nl = lane & 15;
  int wave = blockIdx.x * 4 + w;          // 4096 waves total
  int gh = wave >> 6;
  int i0 = (wave & 63) * 16;
  int g = gh >> 4, h = gh & 15;

  const __bf16* Qp = Qb + (size_t)gh * NNODE * HD;
  const __bf16* Kp = Kb + (size_t)gh * NNODE * HD;
  const __bf16* Vp = Vt + (size_t)gh * HD * NNODE;
  const float* bp = bias + (size_t)gh * NNODE * NNODE;

  union { v16bf v; v8bf h2[2]; } qa;
  const __bf16* qrow = Qp + (size_t)(i0 + nl) * HD;
  qa.h2[0] = *(const v8bf*)(qrow + 16 * half);
  qa.h2[1] = *(const v8bf*)(qrow + 16 * half + 8);

  float mrow[8], lrow[8];
#pragma unroll
  for (int v = 0; v < 8; ++v) { mrow[v] = -__builtin_inff(); lrow[v] = 0.f; }
  v8f O0 = {}, O1 = {};

  for (int j0 = 0; j0 < NNODE; j0 += 32) {
    v16bf kb0 = *(const v16bf*)(Kp + (size_t)(j0 + nl) * HD + 16 * half);
    v16bf kb1 = *(const v16bf*)(Kp + (size_t)(j0 + 16 + nl) * HD + 16 * half);
    v8f zc = {};
    v8f s0 = wmma_bf16(qa.v, kb0, zc);
    v8f s1 = wmma_bf16(qa.v, kb1, zc);

#pragma unroll
    for (int v = 0; v < 8; ++v) {
      const float* br = bp + (size_t)(i0 + v + 8 * half) * NNODE + j0 + nl;
      s0[v] = s0[v] + __builtin_nontemporal_load(br);        // one-shot stream
      s1[v] = s1[v] + __builtin_nontemporal_load(br + 16);
    }

#pragma unroll
    for (int v = 0; v < 8; ++v) {
      float t = fmaxf(s0[v], s1[v]);
      t = fmaxf(t, __shfl_xor(t, 1));
      t = fmaxf(t, __shfl_xor(t, 2));
      t = fmaxf(t, __shfl_xor(t, 4));
      t = fmaxf(t, __shfl_xor(t, 8));     // row max over 16 lanes of half
      float mn = fmaxf(mrow[v], t);
      float c  = __expf(mrow[v] - mn);
      float p0 = __expf(s0[v] - mn);
      float p1 = __expf(s1[v] - mn);
      float rs = p0 + p1;
      rs += __shfl_xor(rs, 1);
      rs += __shfl_xor(rs, 2);
      rs += __shfl_xor(rs, 4);
      rs += __shfl_xor(rs, 8);
      lrow[v] = lrow[v] * c + rs;
      mrow[v] = mn;
      plds[w][v + 8 * half][nl]      = (__bf16)p0;   // C-layout -> LDS
      plds[w][v + 8 * half][16 + nl] = (__bf16)p1;
      O0[v] = O0[v] * c;
      O1[v] = O1[v] * c;
    }

    // reload P as A-fragment (row = lane&15); in-wave DS ops are in-order
    union { v16bf v; v8bf h2[2]; } pa;
    pa.h2[0] = *(const v8bf*)&plds[w][nl][16 * half];
    pa.h2[1] = *(const v8bf*)&plds[w][nl][16 * half + 8];

    v16bf vb0 = *(const v16bf*)(Vp + (size_t)nl * NNODE + j0 + 16 * half);
    v16bf vb1 = *(const v16bf*)(Vp + (size_t)(16 + nl) * NNODE + j0 + 16 * half);
    O0 = wmma_bf16(pa.v, vb0, O0);
    O1 = wmma_bf16(pa.v, vb1, O1);
  }

#pragma unroll
  for (int v = 0; v < 8; ++v) {
    float inv = 1.f / lrow[v];
    int i = i0 + v + 8 * half;
    __bf16* orow = attnbuf + ((size_t)i * NGRAPH + g) * EMBED + h * HD;
    orow[nl]      = (__bf16)(O0[v] * inv);
    orow[16 + nl] = (__bf16)(O1[v] * inv);
  }
}

// ---------------------------------------------------------------------------
// Kernel 3: out = attn @ W_out^T + b_out. M=4096, N=512, K=512, fp32 out.
// ---------------------------------------------------------------------------
__global__ __launch_bounds__(256) void outproj_kernel(
    const __bf16* __restrict__ attnbuf, const float* __restrict__ W_out,
    const float* __restrict__ b_out, float* __restrict__ out) {
  int wave = (blockIdx.x * blockDim.x + threadIdx.x) >> 5;
  int lane = threadIdx.x & 31;
  int half = lane >> 4, nl = lane & 15;
  int mt = wave >> 3, nt = wave & 7;      // 256 x 8 wave tiles
  int m0 = mt * 16, o0 = nt * 64;

  v8f acc[4] = {};
  const __bf16* arow = attnbuf + (size_t)(m0 + nl) * EMBED;
  for (int k0 = 0; k0 < EMBED; k0 += 32) {
    union { v16bf v; v8bf h2[2]; } a;
    a.h2[0] = *(const v8bf*)(arow + k0 + 16 * half);
    a.h2[1] = *(const v8bf*)(arow + k0 + 16 * half + 8);
#pragma unroll
    for (int t = 0; t < 4; ++t) {
      int o = o0 + t * 16 + nl;
      v16bf b = load_b16_f32(W_out + (size_t)o * EMBED + k0 + 16 * half);
      acc[t] = wmma_bf16(a.v, b, acc[t]);
    }
  }
#pragma unroll
  for (int t = 0; t < 4; ++t) {
    int o = o0 + t * 16 + nl;
    float bo = b_out[o];
#pragma unroll
    for (int v = 0; v < 8; ++v) {
      int r = m0 + v + 8 * half;
      __builtin_nontemporal_store(acc[t][v] + bo, out + (size_t)r * EMBED + o);
    }
  }
}

extern "C" void kernel_launch(void* const* d_in, const int* in_sizes, int n_in,
                              void* d_out, int out_size, void* d_ws, size_t ws_size,
                              hipStream_t stream) {
  const float* query = (const float*)d_in[0];   // [1024,4,512]
  const float* bias  = (const float*)d_in[1];   // [64,1024,1024]
  const float* W_in  = (const float*)d_in[2];   // [1536,512]
  const float* b_in  = (const float*)d_in[3];   // [1536]
  const float* W_out = (const float*)d_in[4];   // [512,512]
  const float* b_out = (const float*)d_in[5];   // [512]
  float* out = (float*)d_out;                   // [1024,4,512]

  // Workspace layout (16 MiB total): Q | K | V^T | attn  (4 MiB each, bf16)
  char* ws = (char*)d_ws;
  __bf16* Qb = (__bf16*)(ws);
  __bf16* Kb = (__bf16*)(ws + (4u << 20));
  __bf16* Vt = (__bf16*)(ws + (8u << 20));
  __bf16* attnbuf = (__bf16*)(ws + (12u << 20));

  qkv_kernel<<<768, 256, 0, stream>>>(query, W_in, b_in, Qb, Kb, Vt);
  attn_kernel<<<1024, 128, 0, stream>>>(bias, Qb, Kb, Vt, attnbuf);
  outproj_kernel<<<256, 256, 0, stream>>>(attnbuf, W_out, b_out, out);
}
